// GNNOGBPredictor_69346541962023
// MI455X (gfx1250) — matrix operand order repaired and checked by reading, not compiled, not run on hardware
//
#include <hip/hip_runtime.h>
#include <hip/hip_bf16.h>

// ---------------------------------------------------------------------------
// GNN (GCN + virtual node) for MI455X / gfx1250.
// Node projection, vn-MLP and readout GEMMs run on v_wmma_f32_16x16x32_f16
// with 4 n-tile accumulators per wave and B fragments staged in LDS.
// Edge message pass is fused (edge-proj + relu + norm + scatter-add) with
// We[l] staged in LDS and global_prefetch_b8 on gathered rows.
// ---------------------------------------------------------------------------

#define EPS_BN 1e-5f

typedef __attribute__((ext_vector_type(16))) _Float16 v16h;
typedef __attribute__((ext_vector_type(8)))  _Float16 v8h;
typedef __attribute__((ext_vector_type(8)))  float    v8f;

// ------------------------------ elementwise --------------------------------

__global__ void k_fill_f32(float* __restrict__ p, float v, size_t n) {
  for (size_t i = (size_t)blockIdx.x * blockDim.x + threadIdx.x; i < n;
       i += (size_t)gridDim.x * blockDim.x)
    p[i] = v;
}

__global__ void k_copy_f32(float* __restrict__ d, const float* __restrict__ s, size_t n) {
  for (size_t i = (size_t)blockIdx.x * blockDim.x + threadIdx.x; i < n;
       i += (size_t)gridDim.x * blockDim.x)
    d[i] = s[i];
}

// deg starts at 1.0 (self term); one atomic per edge destination
__global__ void k_edge_deg(float* __restrict__ deg, const int* __restrict__ dst, int E) {
  for (int e = blockIdx.x * blockDim.x + threadIdx.x; e < E;
       e += gridDim.x * blockDim.x)
    atomicAdd(&deg[dst[e]], 1.0f);
}

__global__ void k_norms(const float* __restrict__ deg, float* __restrict__ nrm,
                        float* __restrict__ invdeg, int N) {
  for (int i = blockIdx.x * blockDim.x + threadIdx.x; i < N;
       i += gridDim.x * blockDim.x) {
    float d = deg[i];
    nrm[i] = rsqrtf(d);
    invdeg[i] = 1.0f / d;
  }
}

__global__ void k_gather_emb(float* __restrict__ h, const float* __restrict__ emb,
                             const int* __restrict__ types, int N, int H) {
  size_t total = (size_t)N * H;
  for (size_t i = (size_t)blockIdx.x * blockDim.x + threadIdx.x; i < total;
       i += (size_t)gridDim.x * blockDim.x) {
    size_t r = i / H, c = i % H;
    h[i] = emb[(size_t)types[r] * H + c];
  }
}

__global__ void k_bcast_vn(float* __restrict__ vn, const float* __restrict__ vn_emb,
                           int B, int H) {
  size_t total = (size_t)B * H;
  for (size_t i = (size_t)blockIdx.x * blockDim.x + threadIdx.x; i < total;
       i += (size_t)gridDim.x * blockDim.x)
    vn[i] = vn_emb[i % H];
}

__global__ void k_add_vn(float* __restrict__ h, const float* __restrict__ vn,
                         const int* __restrict__ gid, int N, int H) {
  size_t total = (size_t)N * H;
  for (size_t i = (size_t)blockIdx.x * blockDim.x + threadIdx.x; i < total;
       i += (size_t)gridDim.x * blockDim.x) {
    size_t r = i / H, c = i % H;
    h[i] += vn[(size_t)gid[r] * H + c];
  }
}

// out = BN(agg + relu(x + root)*invdeg) [optional relu]
__global__ void k_combine(float* __restrict__ out, const float* __restrict__ agg,
                          const float* __restrict__ x, const float* __restrict__ root,
                          const float* __restrict__ invdeg,
                          const float* __restrict__ gamma, const float* __restrict__ beta,
                          int N, int H, int dorelu) {
  size_t total = (size_t)N * H;
  float isq = rsqrtf(1.0f + EPS_BN);
  for (size_t i = (size_t)blockIdx.x * blockDim.x + threadIdx.x; i < total;
       i += (size_t)gridDim.x * blockDim.x) {
    size_t r = i / H, c = i % H;
    float v = agg[i] + fmaxf(x[i] + root[c], 0.0f) * invdeg[r];
    v = v * (gamma[c] * isq) + beta[c];
    if (dorelu) v = fmaxf(v, 0.0f);
    out[i] = v;
  }
}

__global__ void k_bn_relu(float* __restrict__ X, const float* __restrict__ g,
                          const float* __restrict__ b, int M, int Nc) {
  size_t total = (size_t)M * Nc;
  float isq = rsqrtf(1.0f + EPS_BN);
  for (size_t i = (size_t)blockIdx.x * blockDim.x + threadIdx.x; i < total;
       i += (size_t)gridDim.x * blockDim.x) {
    size_t c = i % Nc;
    X[i] = fmaxf(X[i] * (g[c] * isq) + b[c], 0.0f);
  }
}

// dst[gid[r]*H + c] += src[r*H + c]   (segment-sum over rows)
__global__ void k_scatter_rows(float* __restrict__ dst, const float* __restrict__ src,
                               const int* __restrict__ gid, int N, int H) {
  size_t total = (size_t)N * H;
  for (size_t i = (size_t)blockIdx.x * blockDim.x + threadIdx.x; i < total;
       i += (size_t)gridDim.x * blockDim.x) {
    size_t r = i / H, c = i % H;
    atomicAdd(&dst[(size_t)gid[r] * H + c], src[i]);
  }
}

__global__ void k_count(float* __restrict__ counts, const int* __restrict__ gid, int N) {
  for (int i = blockIdx.x * blockDim.x + threadIdx.x; i < N;
       i += gridDim.x * blockDim.x)
    atomicAdd(&counts[gid[i]], 1.0f);
}

__global__ void k_gmean(float* __restrict__ gm, const float* __restrict__ gs,
                        const float* __restrict__ counts, int B, int H) {
  size_t total = (size_t)B * H;
  for (size_t i = (size_t)blockIdx.x * blockDim.x + threadIdx.x; i < total;
       i += (size_t)gridDim.x * blockDim.x)
    gm[i] = gs[i] / fmaxf(counts[i / H], 1.0f);
}

// ------------------------------ WMMA packing -------------------------------

// A: f32 (M x K) row-major -> f16 (M x KP) row-major, zero padded
__global__ void k_pack_a(_Float16* __restrict__ Ap, const float* __restrict__ A,
                         int M, int K, int KP) {
  size_t total = (size_t)M * KP;
  for (size_t i = (size_t)blockIdx.x * blockDim.x + threadIdx.x; i < total;
       i += (size_t)gridDim.x * blockDim.x) {
    size_t r = i / KP, c = i % KP;
    Ap[i] = (c < (size_t)K) ? (_Float16)A[r * K + c] : (_Float16)0.0f;
  }
}

// B: f32 W (K x Nc row-major) -> WMMA B fragments (NTt padded to mult of 4).
// Fragment layout: idx = ((nt*KC + kc)*32 + lane)*16 + j
//   lanes 0-15 : col = nt*16 + lane,      halves j -> K = kc*32 + j
//   lanes16-31 : col = nt*16 + (lane-16), halves j -> K = kc*32 + 16 + j
__global__ void k_pack_b(_Float16* __restrict__ Bf, const float* __restrict__ W,
                         int K, int Nc, int KC, int NTt) {
  size_t total = (size_t)NTt * KC * 512;
  for (size_t i = (size_t)blockIdx.x * blockDim.x + threadIdx.x; i < total;
       i += (size_t)gridDim.x * blockDim.x) {
    int j = (int)(i & 15);
    int lane = (int)((i >> 4) & 31);
    size_t rest = i >> 9;
    int kc = (int)(rest % KC);
    int nt = (int)(rest / KC);
    int k = kc * 32 + ((lane < 16) ? 0 : 16) + j;
    int col = nt * 16 + (lane & 15);
    float v = (k < K && col < Nc) ? W[(size_t)k * Nc + col] : 0.0f;
    Bf[i] = (_Float16)v;
  }
}

// ------------------------------ WMMA GEMM ----------------------------------
// C(M,Nc) = A(M,KP f16) * Bfrag + bias.
// Block = 8 waves; each wave: one 16-row m-tile x FOUR 16-col n-tiles.
// The block's 4-n-tile B panel is staged in LDS (ds_load_b128 operands).
__global__ void k_wmma_gemm(const _Float16* __restrict__ A,
                            const _Float16* __restrict__ Bf,
                            const float* __restrict__ bias,
                            float* __restrict__ C,
                            int M, int Nc, int KP, int KC) {
  extern __shared__ __align__(16) _Float16 sB[];   // 4*KC*512 halves
  const int lane = threadIdx.x & 31;
  const int wv = threadIdx.x >> 5;

  // cooperative stage of the 4-tile B panel into LDS (16B per thread per step)
  const size_t panel = (size_t)4 * KC * 512;
  const _Float16* gB = Bf + (size_t)blockIdx.y * panel;
  for (size_t o = (size_t)threadIdx.x * 8; o < panel; o += (size_t)blockDim.x * 8)
    *(v8h*)(sB + o) = *(const v8h*)(gB + o);
  __syncthreads();

  const int mt = blockIdx.x * 8 + wv;
  const int m0 = mt * 16;
  if (m0 >= M) return;                    // wave-uniform exit (after barrier)

  int row = m0 + (lane & 15);
  if (row >= M) row = M - 1;              // keep EXEC all-ones for WMMA
  const _Float16* arow = A + (size_t)row * KP + ((lane < 16) ? 0 : 16);
  const _Float16* sb = sB + lane * 16;

  v8f acc[4] = {v8f{}, v8f{}, v8f{}, v8f{}};
  for (int kc = 0; kc < KC; ++kc) {
    v8h alo = *(const v8h*)(arow + kc * 32);       // K = base+0..7  (or 16..23)
    v8h ahi = *(const v8h*)(arow + kc * 32 + 8);   // K = base+8..15 (or 24..31)
    v16h a;
#pragma unroll
    for (int i = 0; i < 8; ++i) { a[i] = alo[i]; a[i + 8] = ahi[i]; }
#pragma unroll
    for (int t = 0; t < 4; ++t) {
      v16h b = *(const v16h*)(sb + ((size_t)t * KC + kc) * 512);
      acc[t] = __builtin_amdgcn_wmma_f32_16x16x32_f16(
          /*neg_a=*/false, a, /*neg_b=*/false, b,
          /*c_mod=*/(short)0, acc[t], /*reuse_a=*/false, /*reuse_b=*/false);
    }
  }

  const int rbase = m0 + ((lane < 16) ? 0 : 8);
  const int colbase = blockIdx.y * 64 + (lane & 15);
#pragma unroll
  for (int t = 0; t < 4; ++t) {
    int col = colbase + t * 16;
    if (col >= Nc) continue;
    float bv = bias ? bias[col] : 0.0f;
#pragma unroll
    for (int r = 0; r < 8; ++r) {
      int rr = rbase + r;
      if (rr < M) C[(size_t)rr * Nc + col] = acc[t][r] + bv;
    }
  }
}

// ------------------------- fused edge message pass -------------------------
// m = norm[src]*norm[dst] * relu(x[src] + edge_feats @ We + be); agg[dst]+=m
__global__ void k_edge_msg(const int* __restrict__ src, const int* __restrict__ dst,
                           const float* __restrict__ ef, const float* __restrict__ WeL,
                           const float* __restrict__ beL, const float* __restrict__ x,
                           const float* __restrict__ nrm, float* __restrict__ agg,
                           int E, int H) {
  __shared__ float sWe[16 * 300];
  __shared__ float sbe[300];
  for (int i = threadIdx.x; i < 16 * H; i += blockDim.x) sWe[i] = WeL[i];
  for (int i = threadIdx.x; i < H; i += blockDim.x) sbe[i] = beL[i];
  __syncthreads();

  int lane = threadIdx.x & 31;
  int wv = threadIdx.x >> 5;
  for (int e = blockIdx.x * 8 + wv; e < E; e += gridDim.x * 8) {
    int s = src[e], d = dst[e];
    float en = nrm[s] * nrm[d];
    const float* xs = x + (size_t)s * H;
    float* ad = agg + (size_t)d * H;
    __builtin_prefetch(xs + lane, 0, 1);   // global_prefetch_b8 on gathered row
    float f[16];
#pragma unroll
    for (int q = 0; q < 16; ++q) f[q] = ef[(size_t)e * 16 + q];
    for (int c = lane; c < H; c += 32) {
      float acc = sbe[c];
#pragma unroll
      for (int q = 0; q < 16; ++q) acc = fmaf(f[q], sWe[q * H + c], acc);
      float v = xs[c] + acc;
      v = (v > 0.0f) ? v * en : 0.0f;
      atomicAdd(&ad[c], v);
    }
  }
}

// ------------------------------- host side ---------------------------------

static inline int ew_grid(size_t n) {
  size_t b = (n + 255) / 256;
  return (int)(b > 16384 ? 16384 : b);
}

extern "C" void kernel_launch(void* const* d_in, const int* in_sizes, int n_in,
                              void* d_out, int out_size, void* d_ws, size_t ws_size,
                              hipStream_t stream) {
  const int   N = in_sizes[0];
  const int   E = in_sizes[1];
  const int   B = 2048, H = 300, L = 5, EF = 16, T = 128, H2 = 600;
  const int   KP_H = 320, KC_H = 10;                 // K=300 padded
  const int   KP_H2 = 608, KC_H2 = 19;               // K=600 padded
  const int   NTP_H = 20, NTP_H2 = 40, NTP_T = 8;    // col tiles padded to mult 4

  const int*   node_types = (const int*)d_in[0];
  const int*   src        = (const int*)d_in[1];
  const int*   dst        = (const int*)d_in[2];
  const int*   gid        = (const int*)d_in[3];
  const float* edge_feats = (const float*)d_in[4];
  const float* node_emb   = (const float*)d_in[5];
  const float* Wn   = (const float*)d_in[6];
  const float* bn   = (const float*)d_in[7];
  const float* We   = (const float*)d_in[8];
  const float* be   = (const float*)d_in[9];
  const float* root = (const float*)d_in[10];
  const float* gamma= (const float*)d_in[11];
  const float* beta = (const float*)d_in[12];
  const float* vn_emb = (const float*)d_in[13];
  const float* W1 = (const float*)d_in[14];
  const float* b1 = (const float*)d_in[15];
  const float* g1 = (const float*)d_in[16];
  const float* be1= (const float*)d_in[17];
  const float* W2 = (const float*)d_in[18];
  const float* b2 = (const float*)d_in[19];
  const float* g2 = (const float*)d_in[20];
  const float* be2= (const float*)d_in[21];
  const float* pW = (const float*)d_in[22];
  const float* pb = (const float*)d_in[23];
  float* out = (float*)d_out;

  // bump allocator over d_ws
  char* wsb = (char*)d_ws;
  size_t off = 0;
  auto alloc = [&](size_t bytes) -> void* {
    off = (off + 255) & ~(size_t)255;
    void* p = wsb + off;
    off += bytes;
    return p;
  };
  float* deg    = (float*)alloc((size_t)N * 4);
  float* nrm    = (float*)alloc((size_t)N * 4);
  float* invdeg = (float*)alloc((size_t)N * 4);
  float* h      = (float*)alloc((size_t)N * H * 4);
  float* hn     = (float*)alloc((size_t)N * H * 4);
  float* x      = (float*)alloc((size_t)N * H * 4);
  float* agg    = (float*)alloc((size_t)N * H * 4);
  _Float16* hA16   = (_Float16*)alloc((size_t)N * KP_H * 2);
  _Float16* WfragN = (_Float16*)alloc((size_t)NTP_H * KC_H * 512 * 2);
  float* vn   = (float*)alloc((size_t)B * H * 4);
  float* vtmp = (float*)alloc((size_t)B * H * 4);
  float* z    = (float*)alloc((size_t)B * H2 * 4);
  _Float16* vA16   = (_Float16*)alloc((size_t)B * KP_H2 * 2);  // max of KP_H/KP_H2
  _Float16* Wfrag1 = (_Float16*)alloc((size_t)NTP_H2 * KC_H * 512 * 2);
  _Float16* Wfrag2 = (_Float16*)alloc((size_t)NTP_H * KC_H2 * 512 * 2);
  _Float16* WfragP = (_Float16*)alloc((size_t)NTP_T * KC_H * 512 * 2);
  float* counts = (float*)alloc((size_t)B * 4);
  float* gsum   = (float*)alloc((size_t)B * H * 4);
  float* gmean  = (float*)alloc((size_t)B * H * 4);
  (void)ws_size; (void)n_in; (void)out_size;

  const size_t NH = (size_t)N * H;
  const size_t BH = (size_t)B * H;
  const size_t lds10 = (size_t)4 * KC_H * 512 * 2;    // B panel bytes, K=300
  const size_t lds19 = (size_t)4 * KC_H2 * 512 * 2;   // B panel bytes, K=600

  // degree / norms
  k_fill_f32<<<ew_grid(N), 256, 0, stream>>>(deg, 1.0f, (size_t)N);
  k_edge_deg<<<ew_grid(E), 256, 0, stream>>>(deg, dst, E);
  k_norms<<<ew_grid(N), 256, 0, stream>>>(deg, nrm, invdeg, N);

  // h = node_emb[node_types]; vn = broadcast(vn_emb)
  k_gather_emb<<<ew_grid(NH), 256, 0, stream>>>(h, node_emb, node_types, N, H);
  k_bcast_vn<<<ew_grid(BH), 256, 0, stream>>>(vn, vn_emb, B, H);

  float* hcur = h;
  float* hnext = hn;
  const int mblkN = (N + 127) / 128;   // GEMM M blocks (8 m-tiles each)
  const int mblkB = (B + 127) / 128;

  for (int l = 0; l < L; ++l) {
    // h += vn[graph_ids]
    k_add_vn<<<ew_grid(NH), 256, 0, stream>>>(hcur, vn, gid, N, H);

    // x = h @ Wn[l] + bn[l]   (WMMA, 4 n-tiles/wave, B panel in LDS)
    k_pack_a<<<ew_grid((size_t)N * KP_H), 256, 0, stream>>>(hA16, hcur, N, H, KP_H);
    k_pack_b<<<ew_grid((size_t)NTP_H * KC_H * 512), 256, 0, stream>>>(
        WfragN, Wn + (size_t)l * H * H, H, H, KC_H, NTP_H);
    k_wmma_gemm<<<dim3(mblkN, NTP_H / 4), 256, lds10, stream>>>(
        hA16, WfragN, bn + (size_t)l * H, x, N, H, KP_H, KC_H);

    // agg = scatter(edge messages)
    k_fill_f32<<<ew_grid(NH), 256, 0, stream>>>(agg, 0.0f, NH);
    k_edge_msg<<<2048, 256, 0, stream>>>(src, dst, edge_feats,
                                         We + (size_t)l * EF * H, be + (size_t)l * H,
                                         x, nrm, agg, E, H);

    // hnext = BN(agg + relu(x+root)*invdeg) [+relu]
    k_combine<<<ew_grid(NH), 256, 0, stream>>>(
        hnext, agg, x, root + (size_t)l * H, invdeg,
        gamma + (size_t)l * H, beta + (size_t)l * H, N, H, (l != L - 1) ? 1 : 0);

    if (l < L - 1) {
      // vtmp = segment_sum(h_in) + vn ; vn = MLP(vtmp)  (two WMMA GEMMs)
      k_copy_f32<<<ew_grid(BH), 256, 0, stream>>>(vtmp, vn, BH);
      k_scatter_rows<<<ew_grid(NH), 256, 0, stream>>>(vtmp, hcur, gid, N, H);

      k_pack_a<<<ew_grid((size_t)B * KP_H), 256, 0, stream>>>(vA16, vtmp, B, H, KP_H);
      k_pack_b<<<ew_grid((size_t)NTP_H2 * KC_H * 512), 256, 0, stream>>>(
          Wfrag1, W1 + (size_t)l * H * H2, H, H2, KC_H, NTP_H2);
      k_wmma_gemm<<<dim3(mblkB, NTP_H2 / 4), 256, lds10, stream>>>(
          vA16, Wfrag1, b1 + (size_t)l * H2, z, B, H2, KP_H, KC_H);
      k_bn_relu<<<ew_grid((size_t)B * H2), 256, 0, stream>>>(
          z, g1 + (size_t)l * H2, be1 + (size_t)l * H2, B, H2);

      k_pack_a<<<ew_grid((size_t)B * KP_H2), 256, 0, stream>>>(vA16, z, B, H2, KP_H2);
      k_pack_b<<<ew_grid((size_t)NTP_H * KC_H2 * 512), 256, 0, stream>>>(
          Wfrag2, W2 + (size_t)l * H2 * H, H2, H, KC_H2, NTP_H);
      k_wmma_gemm<<<dim3(mblkB, NTP_H / 4), 256, lds19, stream>>>(
          vA16, Wfrag2, b2 + (size_t)l * H, vn, B, H, KP_H2, KC_H2);
      k_bn_relu<<<ew_grid(BH), 256, 0, stream>>>(
          vn, g2 + (size_t)l * H, be2 + (size_t)l * H, B, H);
    }
    // swap h buffers
    float* t = hcur; hcur = hnext; hnext = t;
  }

  // readout: mean over graphs, then linear head (WMMA)
  k_fill_f32<<<ew_grid(B), 256, 0, stream>>>(counts, 0.0f, (size_t)B);
  k_count<<<ew_grid(N), 256, 0, stream>>>(counts, gid, N);
  k_fill_f32<<<ew_grid(BH), 256, 0, stream>>>(gsum, 0.0f, BH);
  k_scatter_rows<<<ew_grid(NH), 256, 0, stream>>>(gsum, hcur, gid, N, H);
  k_gmean<<<ew_grid(BH), 256, 0, stream>>>(gmean, gsum, counts, B, H);

  k_pack_a<<<ew_grid((size_t)B * KP_H), 256, 0, stream>>>(vA16, gmean, B, H, KP_H);
  k_pack_b<<<ew_grid((size_t)NTP_T * KC_H * 512), 256, 0, stream>>>(
      WfragP, pW, H, T, KC_H, NTP_T);
  k_wmma_gemm<<<dim3(mblkB, NTP_T / 4), 256, lds10, stream>>>(
      vA16, WfragP, pb, out, B, T, KP_H, KC_H);
}
